// global_lstmcell_64639257804853
// MI455X (gfx1250) — matrix operand order repaired and checked by reading, not compiled, not run on hardware
//
#include <hip/hip_runtime.h>
#include <hip/hip_bf16.h>

// ---------------------------------------------------------------------------
// Sizes (fixed by the reference problem)
// ---------------------------------------------------------------------------
#define T_STEPS 4096
#define IN_CH   1024
#define KDIM    2048      // LSTM input size = 2*IN_CH (== HID here)
#define HID     2048
#define G4      8192      // 4*HID gate rows

typedef __attribute__((ext_vector_type(16))) __bf16 v16bf;
typedef __attribute__((ext_vector_type(8)))  __bf16 v8bf;
typedef __attribute__((ext_vector_type(8)))  float  v8f;
typedef __attribute__((ext_vector_type(4)))  unsigned int u32x4;
typedef __attribute__((ext_vector_type(8)))  unsigned int u32x8;

// ---------------------------------------------------------------------------
// Workspace layout (bytes). Total ~209 MB.
// ---------------------------------------------------------------------------
static constexpr size_t OFF_XBF  = 0;                                        // T*K bf16   = 16 MB
static constexpr size_t OFF_WIH  = OFF_XBF + (size_t)T_STEPS * KDIM * 2;     // 4H*K bf16  = 32 MB
static constexpr size_t OFF_WHH  = OFF_WIH + (size_t)G4 * KDIM * 2;          // 4H*H bf16  = 32 MB
static constexpr size_t OFF_XG   = OFF_WHH + (size_t)G4 * HID * 2;           // T*4H f32   = 128 MB
static constexpr size_t OFF_BIAS = OFF_XG + (size_t)T_STEPS * G4 * 4;        // 4H f32
static constexpr size_t OFF_HBF  = OFF_BIAS + (size_t)G4 * 4;                // H bf16
static constexpr size_t OFF_SYNC = (OFF_HBF + (size_t)HID * 2 + 255) & ~(size_t)255;

// ---------------------------------------------------------------------------
// Prep kernels: fp32 -> bf16 conversion, concat, bias fuse, barrier reset
// ---------------------------------------------------------------------------
__global__ void lstm_prep_weights(const float* __restrict__ wih, const float* __restrict__ whh,
                                  __bf16* __restrict__ wihb, __bf16* __restrict__ whhb, size_t n) {
  size_t i = (size_t)blockIdx.x * blockDim.x + threadIdx.x;
  if (i < n) {
    wihb[i] = (__bf16)wih[i];
    whhb[i] = (__bf16)whh[i];
  }
}

__global__ void lstm_prep_x(const float* __restrict__ src, const float* __restrict__ dst,
                            __bf16* __restrict__ xb) {
  size_t i = (size_t)blockIdx.x * blockDim.x + threadIdx.x;
  if (i < (size_t)T_STEPS * KDIM) {
    int t = (int)(i / KDIM);
    int c = (int)(i % KDIM);
    float v = (c < IN_CH) ? src[(size_t)t * IN_CH + c]
                          : dst[(size_t)t * IN_CH + (c - IN_CH)];
    xb[i] = (__bf16)v;
  }
}

__global__ void lstm_prep_misc(const float* __restrict__ bih, const float* __restrict__ bhh,
                               float* __restrict__ bias, int* __restrict__ sync) {
  int i = blockIdx.x * blockDim.x + threadIdx.x;
  if (i < G4) bias[i] = bih[i] + bhh[i];
  if (i < 2) sync[i] = 0;   // grid-barrier arrive counter + generation
}

// ---------------------------------------------------------------------------
// Phase 1: xg[T,4H] = x_bf16[T,K] * Wih_bf16[4H,K]^T + bias  (WMMA bf16)
// Block = 256 threads = 8 waves; block tile 64(M) x 128(N); wave tile 16x64.
// ---------------------------------------------------------------------------
__global__ void __launch_bounds__(256)
lstm_gemm_xg(const __bf16* __restrict__ xbf, const __bf16* __restrict__ wih,
             const float* __restrict__ bias, float* __restrict__ xg) {
  const int lane = threadIdx.x & 31;
  const int wave = threadIdx.x >> 5;
  const int wm   = wave & 3;              // M sub-tile (4 x 16 = 64)
  const int wn   = wave >> 2;             // N group   (2 x 64 = 128)
  const int lm   = lane & 15;
  const int lh   = lane >> 4;

  const int tileM = blockIdx.y * 64 + wm * 16;
  const int tileN = blockIdx.x * 128 + wn * 64;

  v8f acc[4] = {};

  const __bf16* arow = xbf + (size_t)(tileM + lm) * KDIM;

  for (int k0 = 0; k0 < KDIM; k0 += 32) {
    // A fragment: lane holds row (tileM+lm), 16 contiguous K at 16*lh
    union { v16bf v; v8bf h[2]; } a;
    a.h[0] = *reinterpret_cast<const v8bf*>(arow + k0 + lh * 16);
    a.h[1] = *reinterpret_cast<const v8bf*>(arow + k0 + lh * 16 + 8);

#pragma unroll
    for (int nt = 0; nt < 4; ++nt) {
      // B fragment: lane holds column n = tileN+nt*16+lm, 16 contiguous K at 16*lh
      const int n = tileN + nt * 16 + lm;
      const v16bf b = *reinterpret_cast<const v16bf*>(wih + (size_t)n * KDIM + k0 + lh * 16);
      acc[nt] = __builtin_amdgcn_wmma_f32_16x16x32_bf16(
          false, a.v, false, b, (short)0, acc[nt], false, false);
    }
  }

#pragma unroll
  for (int nt = 0; nt < 4; ++nt) {
    const int n  = tileN + nt * 16 + lm;
    const float bn = bias[n];
#pragma unroll
    for (int v = 0; v < 8; ++v) {
      const int row = tileM + v + 8 * lh;           // D layout: M = v + 8*(lane>>4)
      xg[(size_t)row * G4 + n] = acc[nt][v] + bn;
    }
  }
}

// ---------------------------------------------------------------------------
// Grid-wide barrier for the persistent recurrent kernel
// ---------------------------------------------------------------------------
#define RBLK 64

__device__ __forceinline__ void grid_barrier(int* cnt, volatile int* gen) {
  __syncthreads();
  if (threadIdx.x == 0) {
    int g = *gen;
    if (atomicAdd(cnt, 1) == RBLK - 1) {
      *cnt = 0;
      __threadfence();
      *gen = g + 1;
    } else {
      while (*gen == g) __builtin_amdgcn_s_sleep(2);
    }
  }
  __syncthreads();
  __threadfence();
}

// ---------------------------------------------------------------------------
// Phase 2: persistent sequential LSTM scan.
// 64 blocks x 256 threads; block b owns h[b*32 .. b*32+31].
// Per step each wave computes a 16-row slice of W_hh*h via WMMA, with h
// broadcast into all 16 B columns (D columns all identical -> no extraction).
// New h is re-staged into LDS each step by the Tensor Data Mover
// (tensor_load_to_lds + s_wait_tensorcnt), issued once by wave 0.
// ---------------------------------------------------------------------------
__global__ void __launch_bounds__(256)
lstm_recurrent(const __bf16* __restrict__ whh, const float* __restrict__ xg,
               __bf16* __restrict__ hbf, float* __restrict__ out,
               int* __restrict__ sync_cnt, int* __restrict__ sync_gen) {
  __shared__ alignas(32) __bf16 hS[HID];   // current h, staged per step
  __shared__ float gS[128];                // 4 gate sections x 32 local h
  __shared__ float cS[32];                 // persistent cell state slice

  const int tid  = threadIdx.x;
  const int lane = tid & 31;
  const int wave = tid >> 5;
  const int lm   = lane & 15;
  const int lh   = lane >> 4;
  const int n0   = blockIdx.x * 32;        // first h index owned by this block
  const int s    = wave >> 1;              // gate section: 0=i 1=f 2=g 3=o
  const int tt   = wave & 1;               // 16-row half of the 32-row slice
  const int rowBase = s * HID + n0 + tt * 16;

  // --- Tensor DMA descriptor (built once; uniform -> SGPRs) ----------------
  // Group 0: count=1 | lds_addr | global_addr(57b) | type=2 (bits 127:126)
  // Group 1: data_size=2B, tensor 2048x1, tile 2048x1, dim0 stride 2048
  const unsigned lds_hS = (unsigned)(unsigned long long)(uintptr_t)&hS[0];
  const unsigned long long ga = (unsigned long long)(uintptr_t)hbf;
  u32x4 tg0;
  tg0[0] = 1u;                                            // count=1
  tg0[1] = lds_hS;                                        // lds_addr (bytes)
  tg0[2] = (unsigned)(ga & 0xFFFFFFFFu);                  // global_addr lo
  tg0[3] = (unsigned)((ga >> 32) & 0x01FFFFFFu) | 0x80000000u; // addr hi | type=2
  u32x8 tg1;
  tg1[0] = 1u << 16;                                      // data_size = 1 (2 bytes)
  tg1[1] = (HID & 0xFFFFu) << 16;                         // tensor_dim0 lo16
  tg1[2] = (HID >> 16) | (1u << 16);                      // tensor_dim0 hi | tensor_dim1=1
  tg1[3] = ((unsigned)HID) << 16;                         // tile_dim0 = 2048
  tg1[4] = 1u;                                            // tile_dim1 = 1
  tg1[5] = (unsigned)HID;                                 // tensor_dim0_stride = 2048
  tg1[6] = 0u;
  tg1[7] = 0u;

  for (int i = tid; i < HID; i += 256) hS[i] = (__bf16)0.0f;
  if (tid < 32) cS[tid] = 0.0f;
  __syncthreads();

  const __bf16* wrow = whh + (size_t)(rowBase + lm) * HID;

  for (int t = 0; t < T_STEPS; ++t) {
    // Two independent accumulator chains over even/odd K-chunks: halves the
    // dependent-WMMA latency of this serial step.
    v8f acc0 = {}, acc1 = {};
    for (int k0 = 0; k0 < HID; k0 += 64) {
      __builtin_prefetch(wrow + k0 + 512, 0, 0);
      union { v16bf v; v8bf h[2]; } a0, a1;
      a0.h[0] = *reinterpret_cast<const v8bf*>(wrow + k0 + lh * 16);
      a0.h[1] = *reinterpret_cast<const v8bf*>(wrow + k0 + lh * 16 + 8);
      a1.h[0] = *reinterpret_cast<const v8bf*>(wrow + k0 + 32 + lh * 16);
      a1.h[1] = *reinterpret_cast<const v8bf*>(wrow + k0 + 32 + lh * 16 + 8);
      const v16bf b0 = *reinterpret_cast<const v16bf*>(&hS[k0 + lh * 16]);
      const v16bf b1 = *reinterpret_cast<const v16bf*>(&hS[k0 + 32 + lh * 16]);
      acc0 = __builtin_amdgcn_wmma_f32_16x16x32_bf16(
          false, a0.v, false, b0, (short)0, acc0, false, false);
      acc1 = __builtin_amdgcn_wmma_f32_16x16x32_bf16(
          false, a1.v, false, b1, (short)0, acc1, false, false);
    }

    // All 16 columns identical; lane lm==0 of each half publishes its 8 rows.
    if (lm == 0) {
#pragma unroll
      for (int v = 0; v < 8; ++v) {
        const int mrow = v + 8 * lh;                 // 0..15 within the 16-row slice
        const int grow = rowBase + mrow;             // global gate row
        gS[s * 32 + tt * 16 + mrow] = acc0[v] + acc1[v] + xg[(size_t)t * G4 + grow];
      }
    }
    __syncthreads();

    if (tid < 32) {
      const float zi = gS[tid];
      const float zf = gS[32 + tid];
      const float zg = gS[64 + tid];
      const float zo = gS[96 + tid];
      const float ig = 1.0f / (1.0f + __expf(-zi));
      const float fg = 1.0f / (1.0f + __expf(-zf));
      const float gg = tanhf(zg);
      const float og = 1.0f / (1.0f + __expf(-zo));
      const float cn = fg * cS[tid] + ig * gg;
      const float hn = og * tanhf(cn);
      cS[tid] = cn;
      hbf[n0 + tid] = (__bf16)hn;
      if (t == T_STEPS - 1) out[n0 + tid] = hn;
    }

    __threadfence();                 // make hbf visible device-wide
    grid_barrier(sync_cnt, sync_gen);

    // Re-stage h (global -> LDS) with the Tensor Data Mover: one DMA op
    // issued by wave 0, completion tracked with TENSORcnt.
    if (wave == 0) {
      asm volatile("tensor_load_to_lds %0, %1"
                   :: "s"(tg0), "s"(tg1)
                   : "memory");
      __builtin_amdgcn_s_wait_tensorcnt(0);
    }
    __syncthreads();
  }
}

// ---------------------------------------------------------------------------
// Launch
// ---------------------------------------------------------------------------
extern "C" void kernel_launch(void* const* d_in, const int* in_sizes, int n_in,
                              void* d_out, int out_size, void* d_ws, size_t ws_size,
                              hipStream_t stream) {
  const float* src = (const float*)d_in[0];
  const float* dst = (const float*)d_in[1];
  const float* Wih = (const float*)d_in[2];
  const float* Whh = (const float*)d_in[3];
  const float* bih = (const float*)d_in[4];
  const float* bhh = (const float*)d_in[5];
  float* out = (float*)d_out;

  char* ws = (char*)d_ws;
  __bf16* xbf  = (__bf16*)(ws + OFF_XBF);
  __bf16* wihb = (__bf16*)(ws + OFF_WIH);
  __bf16* whhb = (__bf16*)(ws + OFF_WHH);
  float*  xg   = (float*)(ws + OFF_XG);
  float*  bias = (float*)(ws + OFF_BIAS);
  __bf16* hbf  = (__bf16*)(ws + OFF_HBF);
  int*    sync = (int*)(ws + OFF_SYNC);

  {
    size_t nw = (size_t)G4 * KDIM;
    lstm_prep_weights<<<(unsigned)((nw + 255) / 256), 256, 0, stream>>>(Wih, Whh, wihb, whhb, nw);
  }
  {
    size_t nx = (size_t)T_STEPS * KDIM;
    lstm_prep_x<<<(unsigned)((nx + 255) / 256), 256, 0, stream>>>(src, dst, xbf);
  }
  lstm_prep_misc<<<(G4 + 255) / 256, 256, 0, stream>>>(bih, bhh, bias, sync);

  dim3 gg(G4 / 128, T_STEPS / 64);
  lstm_gemm_xg<<<gg, 256, 0, stream>>>(xbf, wihb, bias, xg);

  lstm_recurrent<<<RBLK, 256, 0, stream>>>(whhb, xg, hbf, out, sync, sync + 1);
}
